// FusedAttention_74062416053468
// MI455X (gfx1250) — matrix operand order repaired and checked by reading, compiled, not run on hardware
//
#include <hip/hip_runtime.h>

typedef __attribute__((ext_vector_type(16))) _Float16 v16h;
typedef __attribute__((ext_vector_type(8)))  _Float16 v8h;
typedef __attribute__((ext_vector_type(8)))  float    v8f;

typedef int v4i __attribute__((vector_size(16)));
typedef __attribute__((address_space(1))) v4i gv4i;   // global int4
typedef __attribute__((address_space(3))) v4i lv4i;   // LDS int4

#if defined(__has_builtin)
#if __has_builtin(__builtin_amdgcn_global_load_async_to_lds_b128)
#define HAS_ASYNC_LDS 1
#endif
#endif
#ifndef HAS_ASYNC_LDS
#define HAS_ASYNC_LDS 0
#endif

// ---------------------------------------------------------------------------
// 16-byte staging: global f16 -> LDS, async (ASYNCcnt) when available
// ---------------------------------------------------------------------------
__device__ __forceinline__ void stage16(const _Float16* g, _Float16* l) {
#if HAS_ASYNC_LDS
  __builtin_amdgcn_global_load_async_to_lds_b128(
      (gv4i*)(uintptr_t)g, (lv4i*)(uint32_t)(uintptr_t)l, 0, 0);
#else
  *(v8h*)l = *(const v8h*)g;
#endif
}

__device__ __forceinline__ void async_wait0() {
#if HAS_ASYNC_LDS
#if __has_builtin(__builtin_amdgcn_s_wait_asynccnt)
  __builtin_amdgcn_s_wait_asynccnt(0);
#else
  asm volatile("s_wait_asynccnt 0x0" ::: "memory");
#endif
#endif
}

__device__ __forceinline__ void async_wait4() {
#if HAS_ASYNC_LDS
#if __has_builtin(__builtin_amdgcn_s_wait_asynccnt)
  __builtin_amdgcn_s_wait_asynccnt(4);
#else
  asm volatile("s_wait_asynccnt 0x4" ::: "memory");
#endif
#endif
}

// ---------------------------------------------------------------------------
// WMMA helpers (CDNA5 V_WMMA_F32_16X16X32_F16, wave32)
// ---------------------------------------------------------------------------
__device__ __forceinline__ v8f wmma16(v16h a, v16h b, v8f c) {
  return __builtin_amdgcn_wmma_f32_16x16x32_f16(false, a, false, b, (short)0, c,
                                                false, false);
}

// A fragment (16x32 MxK): lane L row M=L%16; chunks at K = kbase+8*(L/16), +16
__device__ __forceinline__ v16h frag_a(const _Float16* base, int stride, int m,
                                       int kbase, int lane) {
  const _Float16* row = base + (m + (lane & 15)) * stride + kbase + 8 * (lane >> 4);
  v8h lo = *(const v8h*)(row);
  v8h hi = *(const v8h*)(row + 16);
  v16h r;
#pragma unroll
  for (int i = 0; i < 8; ++i) { r[i] = lo[i]; r[i + 8] = hi[i]; }
  return r;
}

// B fragment (32x16 KxN) from col-contiguous-K storage: lane L col N=L%16,
// chunks at K = kbase+8*(L/16), +16
__device__ __forceinline__ v16h frag_b(const _Float16* base, int stride, int n,
                                       int kbase, int lane) {
  const _Float16* row = base + (n + (lane & 15)) * stride + kbase + 8 * (lane >> 4);
  v8h lo = *(const v8h*)(row);
  v8h hi = *(const v8h*)(row + 16);
  v16h r;
#pragma unroll
  for (int i = 0; i < 8; ++i) { r[i] = lo[i]; r[i + 8] = hi[i]; }
  return r;
}

// ---------------------------------------------------------------------------
// fp32 -> f16 conversion pre-pass
// ---------------------------------------------------------------------------
__global__ void cvt_f16_k(const float* __restrict__ s, _Float16* __restrict__ d,
                          int n4) {
  for (int i = blockIdx.x * blockDim.x + threadIdx.x; i < n4;
       i += gridDim.x * blockDim.x) {
    float4 v = ((const float4*)s)[i];
    _Float16* p = d + (size_t)i * 4;
    p[0] = (_Float16)v.x; p[1] = (_Float16)v.y;
    p[2] = (_Float16)v.z; p[3] = (_Float16)v.w;
  }
}

// ---------------------------------------------------------------------------
// GEMM: C[M,N](f32) = A[M,K](f16) @ W[N,K](f16)^T
// 128x128 block / 256 threads; wave w: rows (w&3)*32, cols (w>>2)*64
// double-buffered LDS, async staged one K-step ahead
// ---------------------------------------------------------------------------
__global__ __launch_bounds__(256) void gemm_h(const _Float16* __restrict__ A,
                                              const _Float16* __restrict__ W,
                                              float* __restrict__ C,
                                              int M, int N, int K) {
  __shared__ _Float16 As[2][128 * 32];
  __shared__ _Float16 Bs[2][128 * 32];
  const int tid = threadIdx.x, lane = tid & 31, wave = tid >> 5;
  const int mB = blockIdx.y * 128, nB = blockIdx.x * 128;
  const int rp = wave & 3;    // row pair: rows rp*32 .. +31
  const int ch = wave >> 2;   // col half: cols ch*64 .. +63
  v8f acc[2][4] = {};

  // staging: A/B tiles are 128 rows x 32 halves = 512 x 16B chunks each.
  // thread handles A chunks {tid, tid+256} and B chunks {tid, tid+256};
  // chunk c: row = c>>2, half-offset = (c&3)*8  -> 4 async ops / thread / step
  auto issue = [&](int kt, int p) {
#pragma unroll
    for (int i = 0; i < 2; ++i) {
      int c = tid + i * 256, r = c >> 2, o = (c & 3) * 8;
      stage16(A + (size_t)(mB + r) * K + kt + o, &As[p][r * 32 + o]);
      stage16(W + (size_t)(nB + r) * K + kt + o, &Bs[p][r * 32 + o]);
    }
  };

  const int nk = K / 32;
  issue(0, 0);
  for (int t = 0; t < nk; ++t) {
    const int p = t & 1;
    if (t + 1 < nk) issue((t + 1) * 32, p ^ 1);
    if (t + 2 < nk) {  // warm L2 two steps ahead (global_prefetch_b8)
      int c = tid, r = c >> 2, o = (c & 3) * 8;
      __builtin_prefetch(A + (size_t)(mB + r) * K + (t + 2) * 32 + o, 0, 1);
      __builtin_prefetch(W + (size_t)(nB + r) * K + (t + 2) * 32 + o, 0, 1);
    }
    if (t + 1 < nk) async_wait4(); else async_wait0();
    __syncthreads();
    v16h af[2], bf[4];
#pragma unroll
    for (int i = 0; i < 2; ++i) af[i] = frag_a(As[p], 32, rp * 32 + i * 16, 0, lane);
#pragma unroll
    for (int j = 0; j < 4; ++j) bf[j] = frag_b(Bs[p], 32, ch * 64 + j * 16, 0, lane);
#pragma unroll
    for (int i = 0; i < 2; ++i)
#pragma unroll
      for (int j = 0; j < 4; ++j) acc[i][j] = wmma16(af[i], bf[j], acc[i][j]);
    __syncthreads();
  }
#pragma unroll
  for (int i = 0; i < 2; ++i)
#pragma unroll
    for (int j = 0; j < 4; ++j) {
      int rowb = mB + rp * 32 + i * 16 + (lane >> 4) * 8;
      int col  = nB + ch * 64 + j * 16 + (lane & 15);
#pragma unroll
      for (int r = 0; r < 8; ++r)
        C[(size_t)(rowb + r) * N + col] = acc[i][j][r];
    }
}

// ---------------------------------------------------------------------------
// kv_buffer -> kv_new copy
// ---------------------------------------------------------------------------
__global__ void copy_f4(const float4* __restrict__ src, float4* __restrict__ dst,
                        int n4) {
  for (int i = blockIdx.x * blockDim.x + threadIdx.x; i < n4;
       i += gridDim.x * blockDim.x)
    dst[i] = src[i];
}

// ---------------------------------------------------------------------------
// RoPE + scatter + f16 side copies:
//   qh  (2048,32,128) f16, rotated, softmax scale folded
//   kvout fp32 rows (output), kh (2048,8,128) f16 rotated,
//   vth (8,128,2048) f16 transposed V
// ---------------------------------------------------------------------------
__global__ void rope_scatter(const float* __restrict__ cosb,
                             const float* __restrict__ sinb,
                             const float* __restrict__ qbuf,
                             const float* __restrict__ kvp,
                             const int* __restrict__ sel,
                             float* __restrict__ kvout,
                             _Float16* __restrict__ qh,
                             _Float16* __restrict__ kh,
                             _Float16* __restrict__ vth) {
  const int S = 2048, NH = 32, NKV = 8, HD = 128, HH = 64;
  const float scale = 0.08838834764831845f;  // 1/sqrt(128)
  const int stride = gridDim.x * blockDim.x;
  const int t0 = blockIdx.x * blockDim.x + threadIdx.x;
  // q rope -> f16 (scaled)
  for (int i = t0; i < S * NH * HH; i += stride) {
    int d = i % HH, t = i / HH, hh = t % NH, s = t / NH;
    float c0 = cosb[s * HD + d],      s0 = sinb[s * HD + d];
    float c1 = cosb[s * HD + d + HH], s1 = sinb[s * HD + d + HH];
    size_t b = (size_t)s * (NH * HD) + hh * HD + d;
    float x0 = qbuf[b], x1 = qbuf[b + HH];
    qh[b]      = (_Float16)((x0 * c0 - x1 * s0) * scale);
    qh[b + HH] = (_Float16)((x1 * c1 + x0 * s1) * scale);
  }
  // k rope -> kv_new (fp32) + kh (f16)
  for (int i = t0; i < S * NKV * HH; i += stride) {
    int d = i % HH, t = i / HH, hh = t % NKV, s = t / NKV;
    float c0 = cosb[s * HD + d],      s0 = sinb[s * HD + d];
    float c1 = cosb[s * HD + d + HH], s1 = sinb[s * HD + d + HH];
    size_t b = (size_t)s * (2 * NKV * HD) + hh * HD + d;
    float x0 = kvp[b], x1 = kvp[b + HH];
    float k0 = x0 * c0 - x1 * s0, k1 = x1 * c1 + x0 * s1;
    size_t o = (size_t)sel[s] * (2 * NKV * HD) + hh * HD + d;
    kvout[o] = k0; kvout[o + HH] = k1;
    size_t kb = (size_t)s * (NKV * HD) + hh * HD + d;
    kh[kb] = (_Float16)k0; kh[kb + HH] = (_Float16)k1;
  }
  // v -> kv_new (fp32) + vth (f16 transposed [kvh][d][s])
  for (int i = t0; i < S * NKV * HD; i += stride) {
    int d = i % HD, t = i / HD, hh = t % NKV, s = t / NKV;
    float v = kvp[(size_t)s * 2048 + 1024 + hh * HD + d];
    kvout[(size_t)sel[s] * 2048 + 1024 + hh * HD + d] = v;
    vth[(size_t)hh * (HD * S) + (size_t)d * S + s] = (_Float16)v;
  }
}

// ---------------------------------------------------------------------------
// Flash attention: block = (32-query block, head); key blocks of 64.
// All operands pre-converted f16 -> staging is pure async 16B copies.
// ---------------------------------------------------------------------------
__global__ __launch_bounds__(256) void attn_flash(const _Float16* __restrict__ qh,
                                                  const _Float16* __restrict__ kh,
                                                  const _Float16* __restrict__ vth,
                                                  _Float16* __restrict__ attn) {
  __shared__ _Float16 Qs[32 * 136];
  __shared__ _Float16 Ks[64 * 136];
  __shared__ _Float16 Vst[128 * 72];   // [d][key]
  __shared__ float    Ss[32 * 65];
  __shared__ _Float16 Ps[32 * 72];
  __shared__ float    rowm[32], rowl[32], salpha[32];

  const int tid = threadIdx.x, lane = tid & 31, wave = tid >> 5;
  const int qb = blockIdx.x, h = blockIdx.y, kvh = h >> 2;

  // stage Q (32 rows x 128 halves = 512 chunks, 2/thread)
#pragma unroll
  for (int i = 0; i < 2; ++i) {
    int c = tid + i * 256, r = c >> 4, o = (c & 15) * 8;
    stage16(qh + (size_t)(qb * 32 + r) * 4096 + h * 128 + o, &Qs[r * 136 + o]);
  }
  if (tid < 32) { rowm[tid] = -1e30f; rowl[tid] = 0.0f; }

  v8f o0 = {}, o1 = {};
  const int kbmax = (qb * 32 + 31) >> 6;
  for (int kb = 0; kb <= kbmax; ++kb) {
    __syncthreads();
    // stage K rows (1024 chunks) and transposed V (1024 chunks), 8/thread
#pragma unroll
    for (int i = 0; i < 4; ++i) {
      int c = tid + i * 256;
      int r = c >> 4, o = (c & 15) * 8;
      stage16(kh + (size_t)(kb * 64 + r) * 1024 + kvh * 128 + o, &Ks[r * 136 + o]);
      int d = c >> 3, vo = (c & 7) * 8;
      stage16(vth + (size_t)kvh * 262144 + (size_t)d * 2048 + kb * 64 + vo,
              &Vst[d * 72 + vo]);
    }
    async_wait0();
    __syncthreads();
    // S = Q @ K^T  (2x4 tiles; wave w: rg=w&1, cg=w>>1), causal mask
    {
      const int rg = wave & 1, cg = wave >> 1;
      v16h afr[4];
#pragma unroll
      for (int i = 0; i < 4; ++i) afr[i] = frag_a(Qs, 136, rg * 16, i * 32, lane);
      v8f sacc = {};
#pragma unroll
      for (int i = 0; i < 4; ++i)
        sacc = wmma16(afr[i], frag_b(Ks, 136, cg * 16, i * 32, lane), sacc);
      const int rb = rg * 16 + (lane >> 4) * 8;
      const int cc = cg * 16 + (lane & 15);
      const int kcol = kb * 64 + cc;
#pragma unroll
      for (int r = 0; r < 8; ++r)
        Ss[(rb + r) * 65 + cc] = (kcol <= qb * 32 + rb + r) ? sacc[r] : -1e9f;
    }
    __syncthreads();
    // online softmax per row
    if (tid < 32) {
      float m_old = rowm[tid], mx = m_old;
      for (int j = 0; j < 64; ++j) mx = fmaxf(mx, Ss[tid * 65 + j]);
      float alpha = __expf(m_old - mx), lsum = 0.0f;
      for (int j = 0; j < 64; ++j) {
        float p = __expf(Ss[tid * 65 + j] - mx);
        Ps[tid * 72 + j] = (_Float16)p;
        lsum += p;
      }
      rowm[tid] = mx;
      rowl[tid] = rowl[tid] * alpha + lsum;
      salpha[tid] = alpha;
    }
    __syncthreads();
    // O = O*alpha + P @ V  (wave w: rows (w>>2)*16, cols (w&3)*32 + {0,16})
    {
      const int rgO = wave >> 2, cq = wave & 3;
      const int rb = rgO * 16 + (lane >> 4) * 8;
      float arow[8];
#pragma unroll
      for (int r = 0; r < 8; ++r) arow[r] = salpha[rb + r];
#pragma unroll
      for (int r = 0; r < 8; ++r) { o0[r] *= arow[r]; o1[r] *= arow[r]; }
      v16h pa0 = frag_a(Ps, 72, rgO * 16, 0, lane);
      v16h pa1 = frag_a(Ps, 72, rgO * 16, 32, lane);
      o0 = wmma16(pa0, frag_b(Vst, 72, cq * 32,      0,  lane), o0);
      o0 = wmma16(pa1, frag_b(Vst, 72, cq * 32,      32, lane), o0);
      o1 = wmma16(pa0, frag_b(Vst, 72, cq * 32 + 16, 0,  lane), o1);
      o1 = wmma16(pa1, frag_b(Vst, 72, cq * 32 + 16, 32, lane), o1);
    }
  }
  // final 1/l, store f16 attn (S, NH*HD)
  {
    const int rgO = wave >> 2, cq = wave & 3;
    const int rb = rgO * 16 + (lane >> 4) * 8;
    const int dc = cq * 32 + (lane & 15);
#pragma unroll
    for (int r = 0; r < 8; ++r) {
      float linv = 1.0f / rowl[rb + r];
      size_t base = (size_t)(qb * 32 + rb + r) * 4096 + h * 128 + dc;
      attn[base]      = (_Float16)(o0[r] * linv);
      attn[base + 16] = (_Float16)(o1[r] * linv);
    }
  }
}

// ---------------------------------------------------------------------------
// launch
// ---------------------------------------------------------------------------
extern "C" void kernel_launch(void* const* d_in, const int* in_sizes, int n_in,
                              void* d_out, int out_size, void* d_ws, size_t ws_size,
                              hipStream_t stream) {
  const float* x    = (const float*)d_in[0];
  const float* cosb = (const float*)d_in[1];
  const float* sinb = (const float*)d_in[2];
  const float* wq   = (const float*)d_in[3];
  const float* wkv  = (const float*)d_in[4];
  const float* wo   = (const float*)d_in[5];
  const int*   sel  = (const int*)d_in[6];
  const float* kvb  = (const float*)d_in[7];

  float* out   = (float*)d_out;                  // (2048, 4096)
  float* kvout = out + (size_t)2048 * 4096;      // (8192, 16, 128)

  _Float16* xh   = (_Float16*)d_ws;              // 8,388,608 halves
  _Float16* wqh  = xh   + (size_t)8388608;       // 16,777,216
  _Float16* wkvh = wqh  + (size_t)16777216;      // 8,388,608
  _Float16* woh  = wkvh + (size_t)8388608;       // 16,777,216
  _Float16* qh   = woh  + (size_t)16777216;      // 8,388,608
  _Float16* kh   = qh   + (size_t)8388608;       // 2,097,152
  _Float16* vth  = kh   + (size_t)2097152;       // 2,097,152
  _Float16* ath  = vth  + (size_t)2097152;       // 8,388,608
  float* qbuf = (float*)(ath + (size_t)8388608); // 8,388,608 floats
  float* kvp  = qbuf + (size_t)8388608;          // 4,194,304 floats

  dim3 blk(256);
  // fp32 -> f16 pre-pass
  cvt_f16_k<<<2048, 256, 0, stream>>>(x,   xh,   2097152);
  cvt_f16_k<<<4096, 256, 0, stream>>>(wq,  wqh,  4194304);
  cvt_f16_k<<<2048, 256, 0, stream>>>(wkv, wkvh, 2097152);
  cvt_f16_k<<<4096, 256, 0, stream>>>(wo,  woh,  4194304);
  // projections (f16 WMMA, fp32 out)
  gemm_h<<<dim3(32, 16), blk, 0, stream>>>(xh, wqh,  qbuf, 2048, 4096, 4096);
  gemm_h<<<dim3(16, 16), blk, 0, stream>>>(xh, wkvh, kvp,  2048, 2048, 4096);
  // kv cache
  copy_f4<<<4096, 256, 0, stream>>>((const float4*)kvb, (float4*)kvout, 4194304);
  rope_scatter<<<2048, 256, 0, stream>>>(cosb, sinb, qbuf, kvp, sel, kvout,
                                         qh, kh, vth);
  // flash attention -> f16 attn
  attn_flash<<<dim3(64, 32), blk, 0, stream>>>(qh, kh, vth, ath);
  // output projection
  gemm_h<<<dim3(32, 16), blk, 0, stream>>>(ath, woh, out, 2048, 4096, 4096);
}
